// GDPC_67731634258628
// MI455X (gfx1250) — matrix operand order repaired and checked by reading, not compiled, not run on hardware
//
#include <hip/hip_runtime.h>

// CDNA5 / gfx1250 WMMA types
typedef __attribute__((ext_vector_type(16))) _Float16     v16h;
typedef __attribute__((ext_vector_type(8)))  float        v8f;
typedef __attribute__((ext_vector_type(4)))  unsigned int v4u;

#define K1C      256          // filter taps (k+1), compile-time per reference
#define SC       512          // number of series, compile-time per reference
#define M_BLK    256          // rows (t) per workgroup: 8 waves x 32 rows
#define N_BLK    32           // cols (s) per workgroup
#define LDS_STR  264          // fallback path: 256 + 8 halves pad
#define NTHREADS 256

union FragU { v16h v; v4u q[2]; };

// ---------------------------------------------------------------------------
// Preprocess: split fp32 -> (hi, lo) _Float16 so x ~= hi + lo (~22 mantissa
// bits recovered by hi*hi + lo*hi + hi*lo in the WMMA loop). Runs once over
// component (400KB) and beta (512KB); both stay L2-resident afterwards.
// ---------------------------------------------------------------------------
__global__ __launch_bounds__(NTHREADS) void split_hi_lo_kernel(
    const float* __restrict__ src, _Float16* __restrict__ hi,
    _Float16* __restrict__ lo, int n)
{
  const int i = blockIdx.x * NTHREADS + threadIdx.x;
  if (i < n) {
    const float    x = src[i];
    const _Float16 h = (_Float16)x;
    hi[i] = h;
    lo[i] = (_Float16)(x - (float)h);
  }
}

// ---------------------------------------------------------------------------
// Fast path: pure load + WMMA GEMM on pre-split f16 data. No LDS, no
// conversions in the hot loop. 96 v_wmma per wave.
// ---------------------------------------------------------------------------
__global__ __launch_bounds__(NTHREADS) void fits_wmma_pre_kernel(
    const int*      __restrict__ periods,   // [T] int32
    const _Float16* __restrict__ compHi,    // [T + K1C - 1]
    const _Float16* __restrict__ compLo,
    const _Float16* __restrict__ betaHi,    // [SC, K1C] row-major
    const _Float16* __restrict__ betaLo,
    const float*    __restrict__ alpha,     // [SC]
    float*          __restrict__ out,       // [T, SC] row-major
    int T)
{
  const int tid = threadIdx.x;
  const int t0  = blockIdx.x * M_BLK;
  const int s0  = blockIdx.y * N_BLK;

  const int wave = tid >> 5;   // 0..7 -> M sub-block of 32 rows
  const int lane = tid & 31;
  const int lrow = lane & 15;  // A: M row / B,C: N col within 16-tile
  const int lgrp = lane >> 4;  // half-wave group

  // Per-lane gather bases for the two 16-row A subtiles (honors periods[]).
  int pbase[2];
#pragma unroll
  for (int mt = 0; mt < 2; ++mt) {
    const int t = t0 + wave * 32 + mt * 16 + lrow;
    pbase[mt] = (t < T) ? periods[t] : 0;   // p + 255 <= T+254 stays in bounds
  }

  const v8f vzero = {};
  v8f acc[2][2];
#pragma unroll
  for (int mt = 0; mt < 2; ++mt)
#pragma unroll
    for (int nt = 0; nt < 2; ++nt) acc[mt][nt] = vzero;

#pragma unroll
  for (int k0 = 0; k0 < K1C; k0 += 32) {
    // A fragments (16x32 f16): lane M = lrow; K runs lgrp*8+e and 16+lgrp*8+e.
    // Each run of 8 halves is one 16B chunk; run2 sits +32B after run1.
    FragU aH[2], aL[2];
#pragma unroll
    for (int mt = 0; mt < 2; ++mt) {
      const v4u* ph = reinterpret_cast<const v4u*>(compHi + pbase[mt] + k0 + lgrp * 8);
      aH[mt].q[0] = ph[0];
      aH[mt].q[1] = ph[2];          // +16 halves
      const v4u* pl = reinterpret_cast<const v4u*>(compLo + pbase[mt] + k0 + lgrp * 8);
      aL[mt].q[0] = pl[0];
      aL[mt].q[1] = pl[2];
    }

    // B fragments (32x16 f16): lane N = lrow; K runs lgrp*8+e and 16+lgrp*8+e
    // of one beta row = two aligned 16B chunks, +32B apart (L2 / WGP$ resident).
    FragU bH[2], bL[2];
#pragma unroll
    for (int nt = 0; nt < 2; ++nt) {
      const size_t off = (size_t)(s0 + nt * 16 + lrow) * K1C + k0 + lgrp * 8;
      const v4u* ph = reinterpret_cast<const v4u*>(betaHi + off);
      bH[nt].q[0] = ph[0];
      bH[nt].q[1] = ph[2];
      const v4u* pl = reinterpret_cast<const v4u*>(betaLo + off);
      bL[nt].q[0] = pl[0];
      bL[nt].q[1] = pl[2];
    }

    // split-precision MACs: hi*hi + lo*hi + hi*lo, fp32 accumulate
#pragma unroll
    for (int mt = 0; mt < 2; ++mt)
#pragma unroll
      for (int nt = 0; nt < 2; ++nt) {
        acc[mt][nt] = __builtin_amdgcn_wmma_f32_16x16x32_f16(
            false, aH[mt].v, false, bH[nt].v, (short)0, acc[mt][nt], false, false);
        acc[mt][nt] = __builtin_amdgcn_wmma_f32_16x16x32_f16(
            false, aL[mt].v, false, bH[nt].v, (short)0, acc[mt][nt], false, false);
        acc[mt][nt] = __builtin_amdgcn_wmma_f32_16x16x32_f16(
            false, aH[mt].v, false, bL[nt].v, (short)0, acc[mt][nt], false, false);
      }
  }

  // epilogue: C/D layout VGPR r -> M = lgrp*8 + r, N = lrow; add alpha
#pragma unroll
  for (int nt = 0; nt < 2; ++nt) {
    const int s    = s0 + nt * 16 + lrow;
    const float av = alpha[s];
#pragma unroll
    for (int mt = 0; mt < 2; ++mt) {
#pragma unroll
      for (int r = 0; r < 8; ++r) {
        const int t = t0 + wave * 32 + mt * 16 + lgrp * 8 + r;
        if (t < T) out[(size_t)t * SC + s] = acc[mt][nt][r] + av;
      }
    }
  }
}

// ---------------------------------------------------------------------------
// Fallback path (round-1 kernel): converts in-kernel, beta staged in LDS.
// Used only if d_ws is too small for the pre-split arrays.
// ---------------------------------------------------------------------------
__global__ __launch_bounds__(NTHREADS) void fits_wmma_fallback_kernel(
    const int*   __restrict__ periods,
    const float* __restrict__ component,
    const float* __restrict__ beta,
    const float* __restrict__ alpha,
    float*       __restrict__ out,
    int T)
{
  __shared__ _Float16 bHiS[N_BLK * LDS_STR];
  __shared__ _Float16 bLoS[N_BLK * LDS_STR];

  const int tid = threadIdx.x;
  const int t0  = blockIdx.x * M_BLK;
  const int s0  = blockIdx.y * N_BLK;

  for (int i = tid; i < N_BLK * K1C; i += NTHREADS) {
    const int row = i >> 8;
    const int col = i & (K1C - 1);
    const float    x = beta[(size_t)(s0 + row) * K1C + col];
    const _Float16 h = (_Float16)x;
    bHiS[row * LDS_STR + col] = h;
    bLoS[row * LDS_STR + col] = (_Float16)(x - (float)h);
  }
  __syncthreads();

  const int wave = tid >> 5;
  const int lane = tid & 31;
  const int lrow = lane & 15;
  const int lgrp = lane >> 4;

  int pbase[2];
#pragma unroll
  for (int mt = 0; mt < 2; ++mt) {
    const int t = t0 + wave * 32 + mt * 16 + lrow;
    pbase[mt] = (t < T) ? periods[t] : 0;
  }

  const v8f vzero = {};
  v8f acc[2][2];
#pragma unroll
  for (int mt = 0; mt < 2; ++mt)
#pragma unroll
    for (int nt = 0; nt < 2; ++nt) acc[mt][nt] = vzero;

#pragma unroll
  for (int k0 = 0; k0 < K1C; k0 += 32) {
    FragU aH[2], aL[2];
#pragma unroll
    for (int mt = 0; mt < 2; ++mt) {
      const float* src = component + pbase[mt] + k0 + lgrp * 8;
#pragma unroll
      for (int j = 0; j < 8; ++j) {
        const float    x  = src[j];
        const _Float16 hx = (_Float16)x;
        aH[mt].v[j] = hx;
        aL[mt].v[j] = (_Float16)(x - (float)hx);
        const float    y  = src[16 + j];
        const _Float16 hy = (_Float16)y;
        aH[mt].v[8 + j] = hy;
        aL[mt].v[8 + j] = (_Float16)(y - (float)hy);
      }
    }

    FragU bH[2], bL[2];
#pragma unroll
    for (int nt = 0; nt < 2; ++nt) {
      const int off = (nt * 16 + lrow) * LDS_STR + k0 + lgrp * 8;
      const v4u* ph = reinterpret_cast<const v4u*>(&bHiS[off]);
      bH[nt].q[0] = ph[0]; bH[nt].q[1] = ph[2];
      const v4u* pl = reinterpret_cast<const v4u*>(&bLoS[off]);
      bL[nt].q[0] = pl[0]; bL[nt].q[1] = pl[2];
    }

#pragma unroll
    for (int mt = 0; mt < 2; ++mt)
#pragma unroll
      for (int nt = 0; nt < 2; ++nt) {
        acc[mt][nt] = __builtin_amdgcn_wmma_f32_16x16x32_f16(
            false, aH[mt].v, false, bH[nt].v, (short)0, acc[mt][nt], false, false);
        acc[mt][nt] = __builtin_amdgcn_wmma_f32_16x16x32_f16(
            false, aL[mt].v, false, bH[nt].v, (short)0, acc[mt][nt], false, false);
        acc[mt][nt] = __builtin_amdgcn_wmma_f32_16x16x32_f16(
            false, aH[mt].v, false, bL[nt].v, (short)0, acc[mt][nt], false, false);
      }
  }

#pragma unroll
  for (int nt = 0; nt < 2; ++nt) {
    const int s    = s0 + nt * 16 + lrow;
    const float av = alpha[s];
#pragma unroll
    for (int mt = 0; mt < 2; ++mt) {
#pragma unroll
      for (int r = 0; r < 8; ++r) {
        const int t = t0 + wave * 32 + mt * 16 + lgrp * 8 + r;
        if (t < T) out[(size_t)t * SC + s] = acc[mt][nt][r] + av;
      }
    }
  }
}

extern "C" void kernel_launch(void* const* d_in, const int* in_sizes, int n_in,
                              void* d_out, int out_size, void* d_ws, size_t ws_size,
                              hipStream_t stream) {
  const int*   periods   = (const int*)  d_in[0];
  const float* component = (const float*)d_in[1];
  const float* beta      = (const float*)d_in[2];
  const float* alpha     = (const float*)d_in[3];
  float*       out       = (float*)d_out;

  const int T  = in_sizes[0];
  const int nC = in_sizes[1];          // T + K1C - 1
  const int nB = in_sizes[2];          // SC * K1C

  // Workspace layout (16B-aligned halves): compHi, compLo, betaHi, betaLo
  auto align16 = [](size_t x) { return (x + 15) & ~(size_t)15; };
  const size_t szC = align16((size_t)nC * sizeof(_Float16));
  const size_t szB = align16((size_t)nB * sizeof(_Float16));
  const size_t need = 2 * szC + 2 * szB;

  dim3 grid((T + M_BLK - 1) / M_BLK, SC / N_BLK);

  if (ws_size >= need) {
    char* ws = (char*)d_ws;
    _Float16* compHi = (_Float16*)(ws);
    _Float16* compLo = (_Float16*)(ws + szC);
    _Float16* betaHi = (_Float16*)(ws + 2 * szC);
    _Float16* betaLo = (_Float16*)(ws + 2 * szC + szB);

    split_hi_lo_kernel<<<(nC + NTHREADS - 1) / NTHREADS, NTHREADS, 0, stream>>>(
        component, compHi, compLo, nC);
    split_hi_lo_kernel<<<(nB + NTHREADS - 1) / NTHREADS, NTHREADS, 0, stream>>>(
        beta, betaHi, betaLo, nB);
    fits_wmma_pre_kernel<<<grid, NTHREADS, 0, stream>>>(
        periods, compHi, compLo, betaHi, betaLo, alpha, out, T);
  } else {
    fits_wmma_fallback_kernel<<<grid, NTHREADS, 0, stream>>>(
        periods, component, beta, alpha, out, T);
  }
}
